// MorphologicalLSTMForSequenceClassification_81140522156555
// MI455X (gfx1250) — matrix operand order repaired and checked by reading, not compiled, not run on hardware
//
#include <hip/hip_runtime.h>

typedef __attribute__((ext_vector_type(16))) __bf16 v16bf;
typedef __attribute__((ext_vector_type(8)))  float  v8f;

namespace {
constexpr int kB = 32, kT = 4096, kE = 300, kH = 512, kL = 2;
constexpr int NWG  = 16;   // one WG per 32-hidden-unit slice (128 gate rows)
constexpr int NTH  = 256;  // 8 wave32s
constexpr int ROWS = 128;  // gate rows per WG: i,f,g,o x 32
constexpr int JSL  = 32;   // hidden units per WG
constexpr int KTX  = 10;   // ceil(300/32) K-tiles for W_ih (zero padded to 320)
constexpr int KTH  = 16;   // 512/32 K-tiles for W_hh
constexpr int MT   = 8;    // ROWS/16 M-tiles

// LDS fragment pools (element counts, bf16)
constexpr int WIHF = MT * KTX * 512;   // 40960
constexpr int WHHF = MT * KTH * 512;   // 65536
constexpr int XBF  = KTX * 2 * 512;    // 10240
constexpr int HBF  = KTH * 2 * 512;    // 16384
constexpr size_t SMEM_BYTES =
    (size_t)(WIHF + WHHF + XBF + HBF) * 2   // bf16 fragments
    + (size_t)ROWS * kB * 4                 // gates (f32)
    + (size_t)JSL * kB * 4                  // c state (f32)
    + (size_t)ROWS * 4;                     // combined bias (f32)
// = 287,232 B  (< 320 KB CDNA5 WGP LDS)

// workspace layout
constexpr size_t WS_HBF   = 0;                          // 2 x 512 x 32 bf16 (double buffer)
constexpr size_t WS_HF32  = (size_t)2 * kH * kB * 2;    // 512 x 32 f32
constexpr size_t WS_CNT   = WS_HF32 + (size_t)kH * kB * 4;
constexpr size_t WS_BYTES = WS_CNT + 128;
}

__device__ __forceinline__ void grid_barrier(unsigned* cnt, unsigned target) {
  __syncthreads();
  if (threadIdx.x == 0) {
    __threadfence();  // release h writes to device scope
    __hip_atomic_fetch_add(cnt, 1u, __ATOMIC_RELEASE, __HIP_MEMORY_SCOPE_AGENT);
    while (__hip_atomic_load(cnt, __ATOMIC_ACQUIRE, __HIP_MEMORY_SCOPE_AGENT) < target) {
      __builtin_amdgcn_s_sleep(2);
    }
    __threadfence();  // acquire: invalidate near caches before consuming h
  }
  __syncthreads();
}

__global__ __launch_bounds__(NTH, 1)
void lstm_fused_kernel(const float* __restrict__ x,     // [B][T][E]
                       const int*   __restrict__ mask,  // [B][T]
                       const float* __restrict__ Wih,   // [4H][E]
                       const float* __restrict__ Whh,   // [4H][H]
                       const float* __restrict__ bih,
                       const float* __restrict__ bhh,
                       const float* __restrict__ Wcls,  // [L][H]
                       const float* __restrict__ bcls,
                       float* __restrict__ out,         // [B][L]
                       __bf16* __restrict__ hbf,        // [2][H][B] double buffer
                       float*  __restrict__ hf32,       // [H][B]
                       unsigned* __restrict__ cnt)
{
  extern __shared__ __align__(16) char smem[];
  __bf16* wihF  = (__bf16*)smem;          // A-frags of W_ih slice
  __bf16* whhF  = wihF + WIHF;            // A-frags of W_hh slice
  __bf16* xbF   = whhF + WHHF;            // B-frags of x_t   (K=320, N=32)
  __bf16* hbF   = xbF + XBF;              // B-frags of h_t-1 (K=512, N=32)
  float*  gatesL = (float*)(hbF + HBF);   // [128][32]
  float*  cL     = gatesL + ROWS * kB;    // [32][32]
  float*  biasL  = cL + JSL * kB;         // [128]

  const int tid  = threadIdx.x;
  const int lane = tid & 31;
  const int wave = tid >> 5;
  const int wg   = blockIdx.x;

  // ---------------- one-time setup: weights -> bf16 WMMA A-fragments in LDS ---------
  // A-frag (16x32 bf16) per-lane layout: lane l holds row m = l&15;
  // element e (v=e>>1, h=e&1): k = (v>=4?16:0) + (l>=16?8:0) + ((v&3)<<1) + h.
  for (int r = tid; r < ROWS; r += NTH) {
    int g = (r >> 5) * kH + wg * JSL + (r & 31);
    biasL[r] = bih[g] + bhh[g];
  }
  for (int i = tid; i < JSL * kB; i += NTH) cL[i] = 0.f;

  for (int idx = tid; idx < WHHF; idx += NTH) {
    int e = idx & 15, l = (idx >> 4) & 31, fid = idx >> 9;
    int mt = fid / KTH, kt = fid % KTH;
    int mloc = mt * 16 + (l & 15);
    int g = (mloc >> 5) * kH + wg * JSL + (mloc & 31);
    int v = e >> 1;
    int k = ((v >= 4) ? 16 : 0) + ((l >= 16) ? 8 : 0) + ((v & 3) << 1) + (e & 1);
    whhF[idx] = (__bf16)Whh[g * kH + kt * 32 + k];
  }
  for (int idx = tid; idx < WIHF; idx += NTH) {
    int e = idx & 15, l = (idx >> 4) & 31, fid = idx >> 9;
    int mt = fid / KTX, kt = fid % KTX;
    int mloc = mt * 16 + (l & 15);
    int g = (mloc >> 5) * kH + wg * JSL + (mloc & 31);
    int v = e >> 1;
    int k = ((v >= 4) ? 16 : 0) + ((l >= 16) ? 8 : 0) + ((v & 3) << 1) + (e & 1);
    int ee = kt * 32 + k;
    wihF[idx] = (ee < kE) ? (__bf16)Wih[g * kE + ee] : (__bf16)0.f;
  }
  __syncthreads();

  // ---------------- serial time loop ------------------------------------------------
  for (int t = 0; t < kT; ++t) {
    const int cur = t & 1;

    // B-frag (32x16 bf16): lane l holds column n=l&15; element s holds k=(s&7)+8*(l>=16)+16*(s>=8).
    // x_t fragments: one (kt,nt,column) task per thread -> 32 contiguous f32 reads.
    for (int task = tid; task < KTX * 2 * 16; task += NTH) {
      int e = task & 15, nt = (task >> 4) & 1, kt = task >> 5;
      int b = nt * 16 + e;
      const float* xrow = x + ((size_t)b * kT + t) * kE;
      __bf16* dst = xbF + (kt * 2 + nt) * 512;
      int kb = kt * 32;
      #pragma unroll 8
      for (int l = 0; l < 32; ++l) {
        int ed = kb + l;
        dst[(e + ((l & 8) << 1)) * 16 + (l & 7) + ((l & 16) >> 1)] = (ed < kE) ? (__bf16)xrow[ed] : (__bf16)0.f;
      }
    }
    // h_{t-1} fragments: one K-row per task; [j][b] layout -> scatter into B-frag.
    for (int task = tid; task < KTH * 2 * 32; task += NTH) {
      int l = task & 31, nt = (task >> 5) & 1, kt = task >> 6;
      int j = kt * 32 + l;
      const __bf16* s = hbf + (size_t)cur * kH * kB + j * kB + nt * 16;
      __bf16* d = hbF + (kt * 2 + nt) * 512;
      for (int n = 0; n < 16; ++n)
        d[(n + ((l & 8) << 1)) * 16 + (l & 7) + ((l & 16) >> 1)] = s[n];
    }
    __syncthreads();

    // gates = bias + W_ih@x_t + W_hh@h_{t-1}; wave w owns M-tile w, both N-tiles.
    {
      const int mt = wave;
      v8f acc0, acc1;
      const int moff = (lane >= 16) ? 8 : 0;
      #pragma unroll
      for (int e = 0; e < 8; ++e) {
        float bv = biasL[mt * 16 + e + moff];
        acc0[e] = bv;
        acc1[e] = bv;
      }
      #pragma unroll
      for (int kt = 0; kt < KTX; ++kt) {
        v16bf a  = *(const v16bf*)(wihF + (mt * KTX + kt) * 512 + lane * 16);
        v16bf b0 = *(const v16bf*)(xbF + (kt * 2 + 0) * 512 + lane * 16);
        v16bf b1 = *(const v16bf*)(xbF + (kt * 2 + 1) * 512 + lane * 16);
        acc0 = __builtin_amdgcn_wmma_f32_16x16x32_bf16(false, a, false, b0, (short)0, acc0, false, false);
        acc1 = __builtin_amdgcn_wmma_f32_16x16x32_bf16(false, a, false, b1, (short)0, acc1, false, false);
      }
      #pragma unroll
      for (int kt = 0; kt < KTH; ++kt) {
        v16bf a  = *(const v16bf*)(whhF + (mt * KTH + kt) * 512 + lane * 16);
        v16bf b0 = *(const v16bf*)(hbF + (kt * 2 + 0) * 512 + lane * 16);
        v16bf b1 = *(const v16bf*)(hbF + (kt * 2 + 1) * 512 + lane * 16);
        acc0 = __builtin_amdgcn_wmma_f32_16x16x32_bf16(false, a, false, b0, (short)0, acc0, false, false);
        acc1 = __builtin_amdgcn_wmma_f32_16x16x32_bf16(false, a, false, b1, (short)0, acc1, false, false);
      }
      // C-frag layout: VGPR e, lane l -> m = e + 8*(l>=16), n = l&15.
      const int n = lane & 15;
      #pragma unroll
      for (int e = 0; e < 8; ++e) {
        int m = mt * 16 + e + moff;
        gatesL[m * kB + n]      = acc0[e];
        gatesL[m * kB + 16 + n] = acc1[e];
      }
    }
    __syncthreads();

    // elementwise LSTM cell update with mask (fp32 state)
    for (int idx = tid; idx < JSL * kB; idx += NTH) {
      int jl = idx >> 5, b = idx & 31;
      int mv = mask[(size_t)b * kT + t];
      float gi = gatesL[(jl) * kB + b];
      float gf = gatesL[(32 + jl) * kB + b];
      float gg = gatesL[(64 + jl) * kB + b];
      float go = gatesL[(96 + jl) * kB + b];
      float i_ = 1.f / (1.f + expf(-gi));
      float f_ = 1.f / (1.f + expf(-gf));
      float g_ = tanhf(gg);
      float o_ = 1.f / (1.f + expf(-go));
      float cprev = cL[idx];
      float cnew  = f_ * cprev + i_ * g_;
      float hnew  = o_ * tanhf(cnew);
      int jg = wg * JSL + jl;
      float hprev = hf32[jg * kB + b];
      float hs = mv ? hnew : hprev;
      float cs = mv ? cnew : cprev;
      cL[idx] = cs;
      hf32[jg * kB + b] = hs;
      hbf[(size_t)(cur ^ 1) * kH * kB + jg * kB + b] = (__bf16)hs;
    }

    grid_barrier(cnt, (unsigned)(NWG * (t + 1)));
  }

  // ---------------- classifier: logits = h @ W_cls^T + b_cls -----------------------
  if (wg == 0) {
    for (int idx = tid; idx < kB * kL; idx += NTH) {
      int b = idx >> 1, l = idx & 1;
      float acc = bcls[l];
      for (int j = 0; j < kH; ++j) acc += Wcls[l * kH + j] * hf32[j * kB + b];
      out[idx] = acc;  // idx == b*2 + l
    }
  }
}

extern "C" void kernel_launch(void* const* d_in, const int* in_sizes, int n_in,
                              void* d_out, int out_size, void* d_ws, size_t ws_size,
                              hipStream_t stream) {
  (void)in_sizes; (void)n_in; (void)out_size; (void)ws_size;
  const float* x    = (const float*)d_in[0];
  const int*   mask = (const int*)d_in[1];
  const float* Wih  = (const float*)d_in[2];
  const float* Whh  = (const float*)d_in[3];
  const float* bih  = (const float*)d_in[4];
  const float* bhh  = (const float*)d_in[5];
  const float* Wcls = (const float*)d_in[6];
  const float* bcls = (const float*)d_in[7];
  float* out = (float*)d_out;

  __bf16*   hbf  = (__bf16*)((char*)d_ws + WS_HBF);
  float*    hf32 = (float*)((char*)d_ws + WS_HF32);
  unsigned* cnt  = (unsigned*)((char*)d_ws + WS_CNT);

  // zero h0, c handled in-kernel; zero barrier counter (graph-capture safe)
  hipMemsetAsync(d_ws, 0, WS_BYTES, stream);
  hipFuncSetAttribute((const void*)lstm_fused_kernel,
                      hipFuncAttributeMaxDynamicSharedMemorySize, (int)SMEM_BYTES);
  lstm_fused_kernel<<<dim3(NWG), dim3(NTH), SMEM_BYTES, stream>>>(
      x, mask, Wih, Whh, bih, bhh, Wcls, bcls, out, hbf, hf32, cnt);
}
